// QRNNLayer_noncausal_65859028517545
// MI455X (gfx1250) — compile-verified, matrix operand order checked by reading, not run on hardware
//
#include <hip/hip_runtime.h>

// ---------------------------------------------------------------------------
// QRNN (non-causal conv gates + log-space linear scan) for gfx1250 / MI455X
//
// B=8, T=4096, C_IN=256, H=256, K=3.
// Conv as GEMM on the bf16 WMMA path (v_wmma_f32_16x16x32_bf16), x tile staged
// into LDS via async global->LDS copies (ASYNCcnt path), scan as a 3-pass
// chunked log-space scan (the cs[:, -1] term cancels).
// ---------------------------------------------------------------------------

#define Bb   8
#define Tt   4096
#define Cc   256
#define Hh   256
#define Kk   3
#define Ss   (Tt + 1)        // scan length (init row prepended)
#define CHK  64              // scan chunk length
#define NC   ((Ss + CHK - 1) / CHK)   // 65 chunks
#define TW   32              // time rows per conv block

typedef __attribute__((ext_vector_type(16))) __bf16 v16bf;
typedef __attribute__((ext_vector_type(8)))  float  v8f;

typedef unsigned int   u32;
typedef unsigned short u16;

#if defined(__HIP_DEVICE_COMPILE__) && \
    __has_builtin(__builtin_amdgcn_global_load_async_to_lds_b128) && \
    __has_builtin(__builtin_amdgcn_s_wait_asynccnt)
#define HAVE_ASYNC_LDS 1
// exact parameter types per hipcc diagnostic: vector_size(16) int pointee,
// addrspace(1) global source, addrspace(3) LDS destination
typedef int vi4 __attribute__((vector_size(16)));
typedef __attribute__((address_space(1))) vi4 gvi4;
typedef __attribute__((address_space(3))) vi4 svi4;
#else
#define HAVE_ASYNC_LDS 0
#endif

// ---------------- helpers ----------------

__device__ __forceinline__ u16 f32_to_bf16_rne(float f) {
    u32 u = __float_as_uint(f);
    u32 r = (u + 0x7FFFu + ((u >> 16) & 1u)) >> 16;   // round-to-nearest-even
    return (u16)r;
}

__device__ __forceinline__ float log_sigmoid(float x) {
    // stable: min(x,0) - log1p(exp(-|x|))
    float ax = fabsf(x);
    return fminf(x, 0.0f) - log1pf(__expf(-ax));
}

__device__ __forceinline__ float log_add_exp(float a, float b) {
    float m = fmaxf(a, b);
    if (m == -INFINITY) return -INFINITY;
    float n = fminf(a, b);
    return m + log1pf(__expf(n - m));
}

// ---------------- kernel 0: x f32 -> bf16 ----------------

__global__ __launch_bounds__(256) void k_convert_x(const float* __restrict__ x,
                                                   u16* __restrict__ x16, int n) {
    int i = blockIdx.x * 256 + threadIdx.x;
    if (i < n) x16[i] = f32_to_bf16_rne(x[i]);
}

// ---------------- kernel 1: pack weights into B-fragment layout ------------
// B fragment (32x16 bf16, wave32): lane holds column n = lane%16;
// dword j (j<4) holds rows (kbase+2j, kbase+2j+1), dword 4+j holds rows
// (kbase+16+2j, +1), kbase = (lane/16)*8.  Fragment id:
//   f = ((gate*3 + k)*16 + h_tile)*8 + c_chunk
// storage: fragW[f*256 + lane*8 + j]  (u32 dwords)

__global__ __launch_bounds__(256) void k_pack_w(const float* __restrict__ f_w,
                                                const float* __restrict__ z_w,
                                                u32* __restrict__ fragW) {
    int idx = blockIdx.x * 256 + threadIdx.x;            // 768*256 total dwords
    if (idx >= 2 * Kk * 16 * 8 * 256) return;
    int f    = idx >> 8;
    int rem  = idx & 255;
    int lane = rem >> 3;
    int j    = rem & 7;

    int kc = f & 7;          int t1 = f >> 3;
    int nt = t1 & 15;        int t2 = t1 >> 4;
    int k  = t2 % 3;         int g  = t2 / 3;

    int n    = lane & 15;
    int base = (lane >> 4) << 3;
    int jj   = j & 3;
    int half = j >> 2;
    int r0   = base + half * 16 + 2 * jj;
    int c0   = kc * 32 + r0;

    int h = nt * 16 + n;
    const float* src = g ? z_w : f_w;
    float w0 = src[(h * Cc + c0    ) * Kk + k];
    float w1 = src[(h * Cc + c0 + 1) * Kk + k];
    fragW[idx] = (u32)f32_to_bf16_rne(w0) | ((u32)f32_to_bf16_rne(w1) << 16);
}

// ---------------- kernel 2: conv as WMMA GEMM ----------------
// grid.x = B * T/TW blocks; 256 threads = 8 waves.
// Block covers 32 time rows x all 256 H. Wave (th, hq): th = t-half,
// hq = h-quarter (4 tiles of 16 H each), both gates accumulated.

#define LDS_STRIDE 264   // 256 + 8 bf16 pad -> conflict-free ds_read_b128

__global__ __launch_bounds__(256) void k_conv_wmma(const u16* __restrict__ x16,
                                                   const u32* __restrict__ fragW,
                                                   const float* __restrict__ f_b,
                                                   const float* __restrict__ z_b,
                                                   float* __restrict__ rawF,
                                                   float* __restrict__ rawZ) {
    __shared__ u16 ldsA[34 * LDS_STRIDE];

    int blk = blockIdx.x;
    int b   = blk / (Tt / TW);
    int tt  = blk % (Tt / TW);
    int t0  = tt * TW;
    int tid = threadIdx.x;

    // stage x rows [t0-1 .. t0+32] (34 rows x 256 c, bf16) into LDS.
    // Interior rows: async global->LDS (ASYNCcnt path); boundary rows
    // (t=-1, t=T): zero-filled with a plain ds store.
    for (int idx = tid; idx < 34 * 32; idx += 256) {
        int row = idx >> 5;
        int ch  = idx & 31;                       // 8 bf16 (16B) chunks
        int gt  = t0 - 1 + row;
        u16* ldst = &ldsA[row * LDS_STRIDE + ch * 8];
#if HAVE_ASYNC_LDS
        if (gt >= 0 && gt < Tt) {
            __builtin_amdgcn_global_load_async_to_lds_b128(
                (gvi4*)(x16 + ((size_t)b * Tt + gt) * Cc + ch * 8),
                (svi4*)ldst, 0, 0);
        } else {
            *(uint4*)ldst = make_uint4(0u, 0u, 0u, 0u);
        }
#else
        uint4 v = make_uint4(0u, 0u, 0u, 0u);
        if (gt >= 0 && gt < Tt)
            v = *(const uint4*)(x16 + ((size_t)b * Tt + gt) * Cc + ch * 8);
        *(uint4*)ldst = v;
#endif
    }
#if HAVE_ASYNC_LDS
    __builtin_amdgcn_s_wait_asynccnt(0);
#endif
    __syncthreads();

    int wave = tid >> 5, lane = tid & 31;
    int th = wave >> 2, hq = wave & 3;
    int laneR = lane & 15, laneHi = lane >> 4;

    v8f accF[4] = {};
    v8f accZ[4] = {};

    for (int kc = 0; kc < 8; ++kc) {
#pragma unroll
        for (int k = 0; k < 3; ++k) {
            // A fragment: 16x32 bf16 tile (time x c-chunk, shifted by k-1)
            union { uint4 q[2]; v16bf v; } A;
            const u16* ap = &ldsA[(16 * th + k + laneR) * LDS_STRIDE
                                  + kc * 32 + laneHi * 8];
            A.q[0] = *(const uint4*)ap;
            A.q[1] = *(const uint4*)(ap + 16);
#pragma unroll
            for (int j = 0; j < 4; ++j) {
                int nt = hq * 4 + j;
                const u32* wf = fragW + ((size_t)(((0 * 3 + k) * 16 + nt) * 8 + kc) * 256)
                                      + lane * 8;
                const u32* wz = fragW + ((size_t)(((1 * 3 + k) * 16 + nt) * 8 + kc) * 256)
                                      + lane * 8;
                union { uint4 q[2]; v16bf v; } Bf, Bz;
                Bf.q[0] = *(const uint4*)(wf);
                Bf.q[1] = *(const uint4*)(wf + 4);
                Bz.q[0] = *(const uint4*)(wz);
                Bz.q[1] = *(const uint4*)(wz + 4);
                accF[j] = __builtin_amdgcn_wmma_f32_16x16x32_bf16(
                    false, A.v, false, Bf.v, (short)0, accF[j], false, false);
                accZ[j] = __builtin_amdgcn_wmma_f32_16x16x32_bf16(
                    false, A.v, false, Bz.v, (short)0, accZ[j], false, false);
            }
        }
    }

    // epilogue: bias + store (C/D layout: VGPR i -> row base+i, col = lane%16)
#pragma unroll
    for (int j = 0; j < 4; ++j) {
        int nt = hq * 4 + j;
        int h  = nt * 16 + laneR;
        float bf = f_b[h], bz = z_b[h];
        int trow0 = t0 + th * 16 + laneHi * 8;
        size_t base = ((size_t)b * Tt + trow0) * Hh + h;
#pragma unroll
        for (int i = 0; i < 8; ++i) {
            rawF[base + (size_t)i * Hh] = accF[j][i] + bf;
            rawZ[base + (size_t)i * Hh] = accZ[j][i] + bz;
        }
    }
}

// ---------------- scan kernels ----------------
// Per (b,h):  lf=ls(rf), l1mf=ls(-rf), lz=ls(rz)
//   e_s = exclusive local cumsum of lf, L'_s = local cumlogsumexp(lz+l1mf-e)
//   out_s = exp( logaddexp(Mc_chunk, L'_s) + e_s ),  Mc from chunk-prefix pass.

__device__ __forceinline__ void load_gates(const float* rawF, const float* rawZ,
                                           const float* init_f, const float* init_z,
                                           int b, int s, int h,
                                           float& rf, float& rz) {
    if (s == 0) {
        rf = init_f[b * Hh + h];
        rz = init_z[b * Hh + h];
    } else {
        size_t o = ((size_t)b * Tt + (s - 1)) * Hh + h;
        rf = rawF[o];
        rz = rawZ[o];
    }
}

__global__ __launch_bounds__(256) void k_scan_pass1(const float* __restrict__ rawF,
                                                    const float* __restrict__ rawZ,
                                                    const float* __restrict__ init_f,
                                                    const float* __restrict__ init_z,
                                                    float* __restrict__ Stot,
                                                    float* __restrict__ Ltot) {
    int c = blockIdx.x, b = blockIdx.y, h = threadIdx.x;
    int s0 = c * CHK, s1 = min(Ss, s0 + CHK);
    float e = 0.0f, L = -INFINITY;
    for (int s = s0; s < s1; ++s) {
        float rf, rz;
        load_gates(rawF, rawZ, init_f, init_z, b, s, h, rf, rz);
        float lf   = log_sigmoid(rf);
        float l1mf = log_sigmoid(-rf);
        float lz   = log_sigmoid(rz);
        L = log_add_exp(L, lz + l1mf - e);
        e += lf;
    }
    size_t o = ((size_t)b * NC + c) * Hh + h;
    Stot[o] = e;
    Ltot[o] = L;
}

__global__ __launch_bounds__(256) void k_scan_pass2(const float* __restrict__ Stot,
                                                    const float* __restrict__ Ltot,
                                                    float* __restrict__ Mc) {
    int b = blockIdx.x, h = threadIdx.x;
    float Spre = 0.0f, M = -INFINITY;
    for (int c = 0; c < NC; ++c) {
        size_t o = ((size_t)b * NC + c) * Hh + h;
        Mc[o] = (M == -INFINITY) ? -INFINITY : (M + Spre);
        M = log_add_exp(M, Ltot[o] - Spre);
        Spre += Stot[o];
    }
}

__global__ __launch_bounds__(256) void k_scan_pass3(const float* __restrict__ rawF,
                                                    const float* __restrict__ rawZ,
                                                    const float* __restrict__ init_f,
                                                    const float* __restrict__ init_z,
                                                    const float* __restrict__ Mc,
                                                    float* __restrict__ out) {
    int c = blockIdx.x, b = blockIdx.y, h = threadIdx.x;
    int s0 = c * CHK, s1 = min(Ss, s0 + CHK);
    float mc = Mc[((size_t)b * NC + c) * Hh + h];
    float e = 0.0f, L = -INFINITY;
    for (int s = s0; s < s1; ++s) {
        float rf, rz;
        load_gates(rawF, rawZ, init_f, init_z, b, s, h, rf, rz);
        float lf   = log_sigmoid(rf);
        float l1mf = log_sigmoid(-rf);
        float lz   = log_sigmoid(rz);
        L = log_add_exp(L, lz + l1mf - e);
        out[((size_t)b * Ss + s) * Hh + h] = __expf(log_add_exp(mc, L) + e);
        e += lf;
    }
}

// ---------------- launcher ----------------

extern "C" void kernel_launch(void* const* d_in, const int* in_sizes, int n_in,
                              void* d_out, int out_size, void* d_ws, size_t ws_size,
                              hipStream_t stream) {
    const float* x      = (const float*)d_in[0];   // [B,T,C]
    const float* init_f = (const float*)d_in[1];   // [B,H]
    const float* init_z = (const float*)d_in[2];   // [B,H]
    const float* f_w    = (const float*)d_in[3];   // [H,C,K]
    const float* f_b    = (const float*)d_in[4];   // [H]
    const float* z_w    = (const float*)d_in[5];   // [H,C,K]
    const float* z_b    = (const float*)d_in[6];   // [H]
    float* out = (float*)d_out;                    // [B,T+1,H]

    char* ws = (char*)d_ws;
    // workspace layout (byte offsets, 256B aligned)
    u16*   x16   = (u16*)  (ws + 0);                         // 16,777,216 B
    u32*   fragW = (u32*)  (ws + 16777216);                  //    786,432 B
    float* rawF  = (float*)(ws + 17563648);                  // 33,554,432 B
    float* rawZ  = (float*)(ws + 51118080);                  // 33,554,432 B
    float* Stot  = (float*)(ws + 84672512);                  //    532,480 B
    float* Ltot  = (float*)(ws + 85204992);                  //    532,480 B
    float* Mc    = (float*)(ws + 85737472);                  //    532,480 B

    // 0) x -> bf16
    int nX = Bb * Tt * Cc;
    k_convert_x<<<(nX + 255) / 256, 256, 0, stream>>>(x, x16, nX);

    // 1) weights -> WMMA B-fragment order (bf16 pairs packed in dwords)
    int nW = 2 * Kk * 16 * 8 * 256;
    k_pack_w<<<(nW + 255) / 256, 256, 0, stream>>>(f_w, z_w, fragW);

    // 2) conv gates via bf16 WMMA
    k_conv_wmma<<<Bb * (Tt / TW), 256, 0, stream>>>(x16, fragW, f_b, z_b, rawF, rawZ);

    // 3-5) chunked log-space scan
    dim3 gscan(NC, Bb);
    k_scan_pass1<<<gscan, 256, 0, stream>>>(rawF, rawZ, init_f, init_z, Stot, Ltot);
    k_scan_pass2<<<Bb, 256, 0, stream>>>(Stot, Ltot, Mc);
    k_scan_pass3<<<gscan, 256, 0, stream>>>(rawF, rawZ, init_f, init_z, Mc, out);
}
